// DecouplingFlowLayer_60198261621394
// MI455X (gfx1250) — compile-verified, matrix operand order checked by reading, not run on hardware
//
#include <hip/hip_runtime.h>

// ---------------------------------------------------------------------------
// DecouplingFlowLayer for MI455X (gfx1250, wave32, WMMA)
// B=2 S=512 N=256 F=362 D=128
// Haar-DWT (tiny) -> single shared bf16 A-tile GEMM, padded K=384, feeding
// v_wmma_f32_16x16x32_bf16. x is read exactly once (NT hint); outputs are
// streamed with NT stores so the packed weights stay resident in L2/WGP$.
// ---------------------------------------------------------------------------

typedef __attribute__((ext_vector_type(16))) __bf16 v16bf;
typedef __attribute__((ext_vector_type(8)))  float  v8f;
typedef __attribute__((ext_vector_type(2)))  float  f32x2;

#define B_    2
#define S_    512
#define N_    256
#define F_    362
#define D_    128
#define ROWS  (B_ * S_ * N_)          // 262144
#define KPAD  384                     // 362 real + Xl + Xh + zero pad, 12*32
#define KSTEPS 12
#define MTILE 64
#define LDS_ROW_DW 196                // 384 bf16 = 192 dwords, +4 dword pad (bank skew)
#define HALF_OUT ((size_t)ROWS * D_)  // 33554432 floats per projection

// workspace byte offsets
#define WS_XL  0
#define WS_XH  ((size_t)ROWS * 4)
#define WS_WPG ((size_t)2 * ROWS * 4)                       // 2 MiB
#define WS_WPH (WS_WPG + (size_t)D_ * KPAD * 2)             // +96 KiB

__device__ __forceinline__ unsigned short f2bf(float f) {
  unsigned u = __float_as_uint(f);
  u += 0x7FFFu + ((u >> 16) & 1u);    // round-to-nearest-even
  return (unsigned short)(u >> 16);
}

__device__ __forceinline__ unsigned int pack_bf16x2(float lo, float hi) {
  unsigned ulo = __float_as_uint(lo);
  unsigned uhi = __float_as_uint(hi);
  ulo += 0x7FFFu + ((ulo >> 16) & 1u);
  uhi += 0x7FFFu + ((uhi >> 16) & 1u);
  return (ulo >> 16) | (uhi & 0xFFFF0000u);
}

// ---------------------------------------------------------------------------
// Kernel 0: pack weights into padded bf16 [D][KPAD].
//   B'g[d][j] = Wg[d][j-1] for j=1..362   (j=362 multiplies a[362]=Xl)
//   B'h[d][j] = Wh[d][j-1] for j=1..361 ; B'h[d][363] = Wh[d][361] (pairs Xh)
//   everything else 0 (incl. j=0, which pairs the unused raw feature 0)
// ---------------------------------------------------------------------------
__global__ __launch_bounds__(256) void pack_w_kernel(
    const float* __restrict__ wg, const float* __restrict__ wh,
    unsigned short* __restrict__ wpg, unsigned short* __restrict__ wph) {
  int idx = blockIdx.x * 256 + threadIdx.x;
  if (idx >= 2 * D_ * KPAD) return;
  int proj = idx / (D_ * KPAD);
  int rem  = idx - proj * (D_ * KPAD);
  int d = rem / KPAD;
  int j = rem - d * KPAD;
  float v = 0.0f;
  if (proj == 0) {
    if (j >= 1 && j <= 362) v = wg[d * F_ + (j - 1)];
    wpg[rem] = f2bf(v);
  } else {
    if (j >= 1 && j <= 361)      v = wh[d * F_ + (j - 1)];
    else if (j == 363)           v = wh[d * F_ + 361];
    wph[rem] = f2bf(v);
  }
}

// ---------------------------------------------------------------------------
// Kernel 1: Haar DWT of feature 0 along s, then linear interp back to S.
// One block per (b,n) row (512 blocks, 256 threads).
// ---------------------------------------------------------------------------
__global__ __launch_bounds__(256) void dwt_kernel(
    const float* __restrict__ x, float* __restrict__ xl, float* __restrict__ xh) {
  __shared__ float cA[S_ / 2];
  __shared__ float cD[S_ / 2];
  const int m = blockIdx.x;          // b*N + n
  const int b = m >> 8;              // N_=256
  const int n = m & (N_ - 1);
  const int t = threadIdx.x;         // 0..255

  const size_t base = ((size_t)b * S_ * N_ + (size_t)n) * F_;
  const size_t sstride = (size_t)N_ * F_;
  float e = x[base + (size_t)(2 * t)     * sstride];
  float o = x[base + (size_t)(2 * t + 1) * sstride];
  const float is2 = 0.70710678118654752f;
  cA[t] = (e + o) * is2;
  cD[t] = (e - o) * is2;
  __syncthreads();

  for (int t2 = t; t2 < S_; t2 += 256) {
    float src = fmaxf(0.5f * (float)t2 - 0.25f, 0.0f);
    int i0 = (int)src;
    int i1 = min(i0 + 1, S_ / 2 - 1);
    float w = src - (float)i0;
    float vl = cA[i0] * (1.0f - w) + cA[i1] * w;
    float vh = cD[i0] * (1.0f - w) + cD[i1] * w;
    size_t outi = ((size_t)b * S_ + (size_t)t2) * N_ + (size_t)n;  // row index
    xl[outi] = vl;
    xh[outi] = vh;
  }
}

// ---------------------------------------------------------------------------
// Kernel 2: GEMM. Block = 64 rows x 256 outputs (g:128 + h:128).
// 8 waves: wave w -> proj = w>>2 (0:g, 1:h), row-pair = w&1 (32 rows),
// d-half = (w>>1)&1 (64 outputs). Each wave: 2 M-blocks x 4 N-tiles.
// Each B fragment feeds two independent WMMAs (a0xb, a1xb).
// Fragment layout per ISA 7.12.2: lane L: (m or n)=L%16, group g=L/16;
// the 16 halves are K runs [32k+8g..+8) and [32k+8g+16..+8) -> 2x16B loads.
//
// Occupancy strategy: K-loop unrolled only 2x to bound the scheduler's
// prefetch window (~200 live VGPRs) so the waves_per_eu(4) cap (<=256
// VGPRs/wave, 4 waves/SIMD, 2 blocks/WGP) is met WITHOUT spilling.
// (Full unroll -> >256 VGPRs; full unroll + cap -> scratch spills.)
// ---------------------------------------------------------------------------
__global__ __launch_bounds__(256)
__attribute__((amdgpu_waves_per_eu(4))) void gemm_kernel(
    const float* __restrict__ x,
    const float* __restrict__ xl, const float* __restrict__ xh,
    const unsigned short* __restrict__ wpg, const unsigned short* __restrict__ wph,
    const float* __restrict__ bg, const float* __restrict__ bh,
    float* __restrict__ out) {
  __shared__ __align__(16) unsigned int As[MTILE * LDS_ROW_DW];   // 50176 B

  const int rowt = blockIdx.x * MTILE;

  // ---- stage A tile: a[j]=x[row,j] j=0..361, a[362]=Xl, a[363]=Xh, pad 0 ----
  for (int idx = threadIdx.x; idx < MTILE * 192; idx += 256) {
    int rloc = idx / 192;
    int p    = idx - rloc * 192;          // bf16 pair index: j = 2p, 2p+1
    int row  = rowt + rloc;
    float f0, f1;
    if (p < 181) {                        // j = 0..361 from x (8B aligned), NT
      const f32x2 v = __builtin_nontemporal_load(
          (const f32x2*)(x + (size_t)row * F_) + p);
      f0 = v.x; f1 = v.y;
    } else if (p == 181) {                // j = 362,363 -> Xl, Xh
      f0 = xl[row]; f1 = xh[row];
    } else {                              // zero pad
      f0 = 0.0f; f1 = 0.0f;
    }
    As[rloc * LDS_ROW_DW + p] = pack_bf16x2(f0, f1);
  }
  __syncthreads();

  const int wave  = threadIdx.x >> 5;
  const int lane  = threadIdx.x & 31;
  const int proj  = wave >> 2;            // 0 = g, 1 = h
  const int mpair = wave & 1;             // rows [0,32) or [32,64) of tile
  const int nhalf = (wave >> 1) & 1;      // d [0,64) or [64,128)
  const int n16   = lane & 15;
  const int lg    = lane >> 4;

  const uint4* wbase = (const uint4*)(proj ? wph : wpg);  // d*48 uint4 per row
  const float* bias  = proj ? bh : bg;

  union Frag { uint4 u[2]; v16bf v; };

  const unsigned int* arow0 = As + (mpair * 32 + n16) * LDS_ROW_DW + lg * 4;
  const unsigned int* arow1 = arow0 + 16 * LDS_ROW_DW;

  v8f acc[2][4];
  const v8f vzero = {0.f, 0.f, 0.f, 0.f, 0.f, 0.f, 0.f, 0.f};
#pragma unroll
  for (int mb = 0; mb < 2; ++mb)
#pragma unroll
    for (int nt = 0; nt < 4; ++nt) acc[mb][nt] = vzero;

#pragma unroll 2
  for (int kk = 0; kk < KSTEPS; ++kk) {
    Frag a0, a1;
    a0.u[0] = *(const uint4*)(arow0 + kk * 16);       // K = 32kk+8g .. +8
    a0.u[1] = *(const uint4*)(arow0 + kk * 16 + 8);   // K = 32kk+8g+16 .. +8
    a1.u[0] = *(const uint4*)(arow1 + kk * 16);
    a1.u[1] = *(const uint4*)(arow1 + kk * 16 + 8);
#pragma unroll
    for (int nt = 0; nt < 4; ++nt) {
      const int d = nhalf * 64 + nt * 16 + n16;
      const uint4* bp = wbase + (size_t)(d * 48 + kk * 4 + lg);
      Frag bfr;
      bfr.u[0] = bp[0];
      bfr.u[1] = bp[2];
      acc[0][nt] = __builtin_amdgcn_wmma_f32_16x16x32_bf16(
          false, a0.v, false, bfr.v, (short)0, acc[0][nt], false, false);
      acc[1][nt] = __builtin_amdgcn_wmma_f32_16x16x32_bf16(
          false, a1.v, false, bfr.v, (short)0, acc[1][nt], false, false);
    }
  }

  // ---- store: C layout lane L: N=L%16, VGPR v: M = v + 8*(L/16) ----
  float* obase = out + (size_t)proj * HALF_OUT + (size_t)(rowt + mpair * 32) * D_;
#pragma unroll
  for (int mb = 0; mb < 2; ++mb) {
#pragma unroll
    for (int nt = 0; nt < 4; ++nt) {
      const int d = nhalf * 64 + nt * 16 + n16;
      const float bv = bias[d];
#pragma unroll
      for (int v = 0; v < 8; ++v) {
        __builtin_nontemporal_store(
            acc[mb][nt][v] + bv,
            obase + (size_t)(mb * 16 + v + 8 * lg) * D_ + d);
      }
    }
  }
}

// ---------------------------------------------------------------------------
extern "C" void kernel_launch(void* const* d_in, const int* in_sizes, int n_in,
                              void* d_out, int out_size, void* d_ws, size_t ws_size,
                              hipStream_t stream) {
  (void)in_sizes; (void)n_in; (void)out_size; (void)ws_size;
  const float* x    = (const float*)d_in[0];
  const float* Wg_w = (const float*)d_in[1];
  const float* Wg_b = (const float*)d_in[2];
  const float* Wh_w = (const float*)d_in[3];
  const float* Wh_b = (const float*)d_in[4];
  float* out = (float*)d_out;

  char* ws = (char*)d_ws;
  float* Xl = (float*)(ws + WS_XL);
  float* Xh = (float*)(ws + WS_XH);
  unsigned short* Wpg = (unsigned short*)(ws + WS_WPG);
  unsigned short* Wph = (unsigned short*)(ws + WS_WPH);

  hipLaunchKernelGGL(pack_w_kernel, dim3((2 * D_ * KPAD + 255) / 256), dim3(256),
                     0, stream, Wg_w, Wh_w, Wpg, Wph);
  hipLaunchKernelGGL(dwt_kernel, dim3(B_ * N_), dim3(256), 0, stream, x, Xl, Xh);
  hipLaunchKernelGGL(gemm_kernel, dim3(ROWS / MTILE), dim3(256), 0, stream,
                     x, Xl, Xh, Wpg, Wph, Wg_b, Wh_b, out);
}
